// MultiHeadAttention_26285199851609
// MI455X (gfx1250) — compile-verified, matrix operand order checked by reading, not run on hardware
//
#include <hip/hip_runtime.h>
#include <hip/hip_bf16.h>

#define D_MODEL 1024
#define N_HEADS 16
#define D_K     64
#define SEQ     1024
#define BATCH   8

typedef __bf16 bf16_t;
typedef bf16_t v16bf __attribute__((ext_vector_type(16)));
typedef float  v8f   __attribute__((ext_vector_type(8)));

typedef unsigned int u32x4 __attribute__((ext_vector_type(4)));
typedef int          i32x4 __attribute__((ext_vector_type(4)));
typedef int          i32x8 __attribute__((ext_vector_type(8)));

#if defined(__has_builtin)
#if __has_builtin(__builtin_amdgcn_tensor_load_to_lds) && __has_builtin(__builtin_amdgcn_s_wait_tensorcnt)
#define HAVE_TDM 1
#endif
#endif
#ifndef HAVE_TDM
#define HAVE_TDM 0
#endif

__device__ __forceinline__ v8f zero_v8f() {
    v8f z;
#pragma unroll
    for (int i = 0; i < 8; ++i) z[i] = 0.0f;
    return z;
}

__device__ __forceinline__ v8f wmma_bf16(v16bf a, v16bf b, v8f c) {
    // D = A(16x32 bf16) * B(32x16 bf16) + C(16x16 f32)
    return __builtin_amdgcn_wmma_f32_16x16x32_bf16(
        false, a, false, b, (short)0, c, false, false);
}

// ---------------------------------------------------------------------------
// Fragment packing. A WMMA operand tile (16 rows/cols x 32 K, bf16) is stored
// as 32 lanes x 16 contiguous bf16 (512 elems, 1 KB). Element (i, k) with
// i = row (A) or col (B) in 0..15 and k in 0..31 lives at:
//   lane = i + 16*((k>>3)&1)   (ISA 7.12.2: lanes 0-15 hold K 0-7/16-23,
//   e    = (k&7) + 8*((k>>4)&1) lanes 16-31 hold K 8-15/24-31)
// A fragment load is then two contiguous 16B loads per lane (b128).
// ---------------------------------------------------------------------------
__device__ __forceinline__ int frag_off(int i, int k) {
    const int lane = i + (((k >> 3) & 1) << 4);
    const int e    = (k & 7) + (((k >> 4) & 1) << 3);
    return lane * 16 + e;
}

__device__ __forceinline__ v16bf load_frag_packed(const bf16_t* tile) {
    const int lane = threadIdx.x & 31;
    const uint4* p = (const uint4*)(tile + lane * 16);   // 32B per lane
    union { uint4 u[2]; v16bf v; } cvt;
    cvt.u[0] = p[0];
    cvt.u[1] = p[1];
    return cvt.v;
}

// A fragment from a row-major bf16 tile (row stride ld elements) — used for
// LDS-resident matrices produced row-major (softmax probs).
__device__ __forceinline__ v16bf load_a_frag_rm(const bf16_t* p, int ld) {
    const int lane = threadIdx.x & 31;
    const int row  = lane & 15;
    const int kb   = (lane < 16) ? 0 : 8;
    const bf16_t* r = p + row * ld + kb;
    v16bf a;
#pragma unroll
    for (int e = 0; e < 8; ++e) a[e] = r[e];
#pragma unroll
    for (int e = 0; e < 8; ++e) a[8 + e] = r[16 + e];
    return a;
}

// B fragment from f32 weight matrix accessed transposed: B[k][n] = W[n*ld + k]
__device__ __forceinline__ v16bf load_b_frag_f32t(const float* p, int ld) {
    const int lane = threadIdx.x & 31;
    const int n    = lane & 15;
    const int kb   = (lane < 16) ? 0 : 8;
    const float* r = p + (size_t)n * ld + kb;
    v16bf b;
#pragma unroll
    for (int e = 0; e < 8; ++e) b[e] = (bf16_t)r[e];
#pragma unroll
    for (int e = 0; e < 8; ++e) b[8 + e] = (bf16_t)r[16 + e];
    return b;
}

// ---------------------------------------------------------------------------
// TDM: 1-D contiguous bf16 copy global -> LDS via tensor_load_to_lds.
// D# per ISA ch.8: group0 = {count=1 | type=2, lds_addr, global_addr},
// group1 = {data_size=2B, tensor_dim0=tile_dim0=nelem, dim1=1, stride0=nelem}.
// Issued by one wave; completion via s_wait_tensorcnt 0 then barrier.
// ---------------------------------------------------------------------------
__device__ __forceinline__ void tdm_load_1d_bf16(const void* gsrc, unsigned lds_off,
                                                 unsigned nelem) {
#if HAVE_TDM
    const unsigned long long ga = (unsigned long long)(size_t)gsrc;
    u32x4 g0;
    g0[0] = 1u;                                            // count=1, user mode
    g0[1] = lds_off;                                       // lds_addr (bytes)
    g0[2] = (unsigned)(ga & 0xFFFFFFFFu);                  // global_addr[31:0]
    g0[3] = (unsigned)((ga >> 32) & 0x01FFFFFFu) | (2u << 30);  // [56:32] | type=2
    i32x8 g1;
    g1[0] = 1 << 16;                                       // data_size=1 (2 bytes)
    g1[1] = (int)((nelem & 0xFFFFu) << 16);                // tensor_dim0[15:0]
    g1[2] = (int)((nelem >> 16) & 0xFFFFu) | (1 << 16);    // tensor_dim0[31:16], dim1=1
    g1[3] = (int)((nelem & 0xFFFFu) << 16);                // tile_dim0 = nelem
    g1[4] = 0;                                             // tile_dim1/2 unused
    g1[5] = (int)nelem;                                    // tensor_dim0_stride[31:0]
    g1[6] = 0;
    g1[7] = 0;
    i32x4 g2 = {0, 0, 0, 0};
    i32x4 g3 = {0, 0, 0, 0};
#if __clang_major__ >= 23
    i32x8 g4 = {0, 0, 0, 0, 0, 0, 0, 0};
    __builtin_amdgcn_tensor_load_to_lds(g0, g1, g2, g3, g4, 0);
#else
    __builtin_amdgcn_tensor_load_to_lds(g0, g1, g2, g3, 0);
#endif
    __builtin_amdgcn_s_wait_tensorcnt(0);
#else
    (void)gsrc; (void)lds_off; (void)nelem;
#endif
}

// ---------------------------------------------------------------------------
// Kernel 1: Q/K/V projections (blockIdx.z selects). Block tile 64x64,
// 128 threads / 4 waves, K loop 32. Outputs are written FRAGMENT-PACKED bf16:
//   Qp : per bh, A-tiles [sTile(64)][kt(2)]        (scores A operand)
//   Kt : per bh, B-tiles [kt(2)][nTile(64)]        (scores B operand, K^T)
//   Vp : per bh, B-tiles [kt(32)][nTile(4)]        (context B operand)
// ---------------------------------------------------------------------------
__global__ void proj_kernel(const float* __restrict__ Xq, const float* __restrict__ Xk,
                            const float* __restrict__ Xv,
                            const float* __restrict__ Wq, const float* __restrict__ Wk,
                            const float* __restrict__ Wv,
                            bf16_t* __restrict__ Qp, bf16_t* __restrict__ Kt,
                            bf16_t* __restrict__ Vp) {
    const int which = blockIdx.z;
    const float* X = (which == 0) ? Xq : ((which == 1) ? Xk : Xv);
    const float* W = (which == 0) ? Wq : ((which == 1) ? Wk : Wv);

    const int ALD = 36;
    __shared__ bf16_t As[64 * 36];

    const int m0   = blockIdx.x * 64;
    const int n0   = blockIdx.y * 64;
    const int tid  = threadIdx.x;
    const int wave = tid >> 5;
    const int wr   = wave >> 1;
    const int wc   = wave & 1;

    v8f acc[2][2];
#pragma unroll
    for (int i = 0; i < 2; ++i)
#pragma unroll
        for (int j = 0; j < 2; ++j) acc[i][j] = zero_v8f();

    for (int kk = 0; kk < D_MODEL; kk += 32) {
        for (int i = tid; i < 64 * 32; i += 128) {
            const int r = i >> 5, c = i & 31;
            As[r * ALD + c] = (bf16_t)X[(size_t)(m0 + r) * D_MODEL + kk + c];
        }
        if (kk + 32 < D_MODEL)
            __builtin_prefetch(&X[(size_t)(m0 + (tid >> 1)) * D_MODEL + kk + 32], 0, 1);
        __syncthreads();

        const v16bf a0 = load_a_frag_rm(As + (wr * 32 + 0)  * ALD, ALD);
        const v16bf a1 = load_a_frag_rm(As + (wr * 32 + 16) * ALD, ALD);
#pragma unroll
        for (int tj = 0; tj < 2; ++tj) {
            const v16bf b = load_b_frag_f32t(
                W + (size_t)(n0 + wc * 32 + tj * 16) * D_MODEL + kk, D_MODEL);
            acc[0][tj] = wmma_bf16(a0, b, acc[0][tj]);
            acc[1][tj] = wmma_bf16(a1, b, acc[1][tj]);
        }
        __syncthreads();
    }

    const int lane  = tid & 31;
    const int rofs  = (lane < 16) ? 0 : 8;
    const int nlane = lane & 15;
#pragma unroll
    for (int ti = 0; ti < 2; ++ti)
#pragma unroll
        for (int tj = 0; tj < 2; ++tj)
#pragma unroll
            for (int r = 0; r < 8; ++r) {
                const int m = m0 + wr * 32 + ti * 16 + r + rofs;  // b*S + s
                const int o = n0 + wc * 32 + tj * 16 + nlane;     // h*64 + d
                const int b = m >> 10, s = m & 1023;
                const int h = o >> 6,  d = o & 63;
                const int bh = b * N_HEADS + h;
                const bf16_t v = (bf16_t)acc[ti][tj][r];
                if (which == 0) {
                    // A-tile: i=row-in-tile (s&15), k=d&31; tiles [s>>4][d>>5]
                    const size_t t = ((size_t)bh * 64 + (s >> 4)) * 2 + (d >> 5);
                    Qp[t * 512 + frag_off(s & 15, d & 31)] = v;
                } else if (which == 1) {
                    // B-tile (K^T): i=col (s&15), k=d&31; tiles [d>>5][s>>4]
                    const size_t t = ((size_t)bh * 2 + (d >> 5)) * 64 + (s >> 4);
                    Kt[t * 512 + frag_off(s & 15, d & 31)] = v;
                } else {
                    // B-tile (V): i=col (d&15), k=s&31; tiles [s>>5][d>>4]
                    const size_t t = ((size_t)bh * 32 + (s >> 5)) * 4 + (d >> 4);
                    Vp[t * 512 + frag_off(d & 15, s & 31)] = v;
                }
            }
}

// ---------------------------------------------------------------------------
// Kernel 2: per (b,h, 32-query strip): scores -> LDS (128 KB), mask+scale,
// wave-per-row softmax, f32 attn out + bf16 probs, context = P*V.
// Q tile arrives via TDM (tensor_load_to_lds) as 4 packed A-fragments.
// ---------------------------------------------------------------------------
__global__ void attn_kernel(const bf16_t* __restrict__ Qp, const bf16_t* __restrict__ Kt,
                            const bf16_t* __restrict__ Vp,
                            const unsigned char* __restrict__ mask,
                            float* __restrict__ attn_out, bf16_t* __restrict__ ctx) {
    extern __shared__ char smem[];
    float*  sc = (float*)smem;                                    // 32 x 1024 f32
    bf16_t* pb = (bf16_t*)(smem + 32 * SEQ * 4);                  // 32 x 1024 bf16
    bf16_t* qa = (bf16_t*)(smem + 32 * SEQ * 4 + 32 * SEQ * 2);   // 4 packed frags

    const int q0   = blockIdx.x * 32;
    const int bh   = blockIdx.y;
    const int b    = bh >> 4;
    const int h    = bh & 15;
    const int tid  = threadIdx.x;
    const int wave = tid >> 5;
    const int lane = tid & 31;
    const int rofs = (lane < 16) ? 0 : 8;
    const int nlane = lane & 15;

    // stage Q strip: contiguous 2048 packed bf16 (tiles [q0/16..+1][kt 0..1])
    const bf16_t* Qsrc = Qp + ((size_t)bh * 64 + (q0 >> 4)) * 2 * 512;
#if HAVE_TDM
    if (tid == 0)
        tdm_load_1d_bf16(Qsrc, (unsigned)(size_t)(void*)qa, 2048);
#else
    for (int i = tid; i < 2048; i += 256) qa[i] = Qsrc[i];
#endif
    __syncthreads();

    // ---- scores: 2 x 64 grid of 16x16 tiles over [32 q] x [1024 k] ----
    const bf16_t* Ktb = Kt + (size_t)bh * (2 * 64 * 512);
    for (int t = wave; t < 128; t += 8) {
        const int rt = t >> 6;
        const int ct = t & 63;
        v8f acc = zero_v8f();
#pragma unroll
        for (int kt = 0; kt < 2; ++kt) {
            const v16bf a  = load_frag_packed(qa + (rt * 2 + kt) * 512);
            const v16bf bb = load_frag_packed(Ktb + ((size_t)kt * 64 + ct) * 512);
            acc = wmma_bf16(a, bb, acc);
        }
#pragma unroll
        for (int r = 0; r < 8; ++r) {
            const int row = rt * 16 + r + rofs;
            const int col = ct * 16 + nlane;
            float v = acc[r] * 0.125f;  // 1/sqrt(64)
            if (mask[((size_t)b * SEQ + q0 + row) * SEQ + col]) v = -1.0e9f;
            sc[row * SEQ + col] = v;
        }
    }
    __syncthreads();

    // ---- softmax: each wave owns 4 rows ----
    float* attn_base = attn_out + (size_t)bh * SEQ * SEQ;
#pragma unroll
    for (int i = 0; i < 4; ++i) {
        const int row = wave * 4 + i;
        float* srow = sc + row * SEQ;
        float mx = -3.0e38f;
        for (int c = lane; c < SEQ; c += 32) mx = fmaxf(mx, srow[c]);
#pragma unroll
        for (int m = 16; m > 0; m >>= 1) mx = fmaxf(mx, __shfl_xor(mx, m, 32));
        float sum = 0.0f;
        for (int c = lane; c < SEQ; c += 32) {
            const float e = __expf(srow[c] - mx);
            srow[c] = e;
            sum += e;
        }
#pragma unroll
        for (int m = 16; m > 0; m >>= 1) sum += __shfl_xor(sum, m, 32);
        const float inv = 1.0f / sum;
        float*  arow = attn_base + (size_t)(q0 + row) * SEQ;
        bf16_t* prow = pb + row * SEQ;
        for (int c = lane; c < SEQ; c += 32) {
            const float p = srow[c] * inv;
            arow[c] = p;
            prow[c] = (bf16_t)p;
        }
    }
    __syncthreads();

    // ---- context = P(32x1024)*V(1024x64): 2x4 tiles, one per wave ----
    const bf16_t* Vb = Vp + (size_t)bh * (32 * 4 * 512);
    const int rt = wave >> 2;
    const int cf = wave & 3;
    v8f acc = zero_v8f();
    for (int kt = 0; kt < 32; ++kt) {
        const v16bf a  = load_a_frag_rm(pb + (rt * 16) * SEQ + kt * 32, SEQ);
        const v16bf bb = load_frag_packed(Vb + ((size_t)kt * 4 + cf) * 512);
        acc = wmma_bf16(a, bb, acc);
    }
#pragma unroll
    for (int r = 0; r < 8; ++r) {
        const int m = b * SEQ + q0 + rt * 16 + r + rofs;     // global row b*S+s
        const int c = h * D_K + cf * 16 + nlane;             // feature col
        // ctx stored as packed A-tiles [m>>4][c>>5] for the fc GEMM
        const size_t t = ((size_t)(m >> 4)) * 32 + (c >> 5);
        ctx[t * 512 + frag_off(m & 15, c & 31)] = (bf16_t)acc[r];
    }
}

// ---------------------------------------------------------------------------
// Kernel 3: fc GEMM (32-row strip via TDM into LDS) + residual + LayerNorm.
// ---------------------------------------------------------------------------
__global__ void fc_ln_kernel(const bf16_t* __restrict__ ctx, const float* __restrict__ Wfc,
                             const float* __restrict__ inQ, const float* __restrict__ gamma,
                             const float* __restrict__ beta, float* __restrict__ out) {
    extern __shared__ char smem[];
    bf16_t* Cs = (bf16_t*)smem;                      // 2x32 packed A-tiles (64 KB)
    float*  Os = (float*)(smem + 32 * D_MODEL * 2);  // 32 x 1024 f32 (128 KB)

    const int m0   = blockIdx.x * 32;
    const int tid  = threadIdx.x;
    const int wave = tid >> 5;
    const int lane = tid & 31;
    const int rofs = (lane < 16) ? 0 : 8;
    const int nlane = lane & 15;

    // ctx strip: contiguous 32768 packed bf16 (mTiles m0/16..+1, kt 0..31)
    const bf16_t* Csrc = ctx + ((size_t)(m0 >> 4)) * 32 * 512;
#if HAVE_TDM
    if (tid == 0)
        tdm_load_1d_bf16(Csrc, (unsigned)(size_t)(void*)Cs, 32768);
#else
    for (int i = tid; i < 32768; i += 256) Cs[i] = Csrc[i];
#endif
    __syncthreads();

    for (int t = wave; t < 128; t += 8) {
        const int rt = t >> 6;
        const int ct = t & 63;
        v8f acc = zero_v8f();
        for (int kt = 0; kt < 32; ++kt) {
            const v16bf a  = load_frag_packed(Cs + ((size_t)rt * 32 + kt) * 512);
            const v16bf bb = load_b_frag_f32t(
                Wfc + (size_t)(ct * 16) * D_MODEL + kt * 32, D_MODEL);
            acc = wmma_bf16(a, bb, acc);
        }
#pragma unroll
        for (int r = 0; r < 8; ++r) {
            const int row = rt * 16 + r + rofs;
            const int col = ct * 16 + nlane;
            Os[row * D_MODEL + col] = acc[r];
        }
    }
    __syncthreads();

    // residual + LayerNorm: each wave owns 4 rows
#pragma unroll
    for (int i = 0; i < 4; ++i) {
        const int row = wave * 4 + i;
        float* orow = Os + row * D_MODEL;
        const float* qrow = inQ + (size_t)(m0 + row) * D_MODEL;
        float sum = 0.0f, sq = 0.0f;
        for (int c = lane; c < D_MODEL; c += 32) {
            const float v = orow[c] + qrow[c];
            orow[c] = v;
            sum += v;
            sq  += v * v;
        }
#pragma unroll
        for (int m = 16; m > 0; m >>= 1) {
            sum += __shfl_xor(sum, m, 32);
            sq  += __shfl_xor(sq,  m, 32);
        }
        const float mean = sum * (1.0f / (float)D_MODEL);
        const float var  = sq  * (1.0f / (float)D_MODEL) - mean * mean;
        const float rstd = rsqrtf(var + 1e-5f);
        float* out_row = out + (size_t)(m0 + row) * D_MODEL;
        for (int c = lane; c < D_MODEL; c += 32)
            out_row[c] = (orow[c] - mean) * rstd * gamma[c] + beta[c];
    }
}

extern "C" void kernel_launch(void* const* d_in, const int* in_sizes, int n_in,
                              void* d_out, int out_size, void* d_ws, size_t ws_size,
                              hipStream_t stream) {
    (void)in_sizes; (void)n_in; (void)out_size; (void)ws_size;

    const float* inQ  = (const float*)d_in[0];
    const float* inK  = (const float*)d_in[1];
    const float* inV  = (const float*)d_in[2];
    const unsigned char* mask = (const unsigned char*)d_in[3];
    const float* Wq   = (const float*)d_in[4];
    const float* Wk   = (const float*)d_in[5];
    const float* Wv   = (const float*)d_in[6];
    const float* Wfc  = (const float*)d_in[7];
    const float* gma  = (const float*)d_in[8];
    const float* bta  = (const float*)d_in[9];

    float* out  = (float*)d_out;                                   // [B,S,D]
    float* attn = out + (size_t)BATCH * SEQ * D_MODEL;             // [B,H,S,S]

    const size_t per = (size_t)BATCH * N_HEADS * SEQ * D_K;        // 8 Mi elems
    bf16_t* Qp  = (bf16_t*)d_ws;
    bf16_t* Kt  = Qp + per;
    bf16_t* Vp  = Kt + per;
    bf16_t* ctx = Vp + per;

    proj_kernel<<<dim3((BATCH * SEQ) / 64, D_MODEL / 64, 3), 128, 0, stream>>>(
        inQ, inK, inV, Wq, Wk, Wv, Qp, Kt, Vp);

    const size_t sh_attn = (size_t)32 * SEQ * 4 + 32 * SEQ * 2 + 4 * 512 * 2; // 200704 B
    hipFuncSetAttribute((const void*)attn_kernel,
                        hipFuncAttributeMaxDynamicSharedMemorySize, (int)sh_attn);
    attn_kernel<<<dim3(SEQ / 32, BATCH * N_HEADS), 256, sh_attn, stream>>>(
        Qp, Kt, Vp, mask, attn, ctx);

    const size_t sh_fc = (size_t)32 * D_MODEL * 2 + 32 * D_MODEL * 4;         // 196608 B
    hipFuncSetAttribute((const void*)fc_ln_kernel,
                        hipFuncAttributeMaxDynamicSharedMemorySize, (int)sh_fc);
    fc_ln_kernel<<<dim3((BATCH * SEQ) / 32), 256, sh_fc, stream>>>(
        ctx, Wfc, inQ, gma, bta, out);
}
